// GAT_76312978915997
// MI455X (gfx1250) — matrix operand order repaired and checked by reading, not compiled, not run on hardware
//
#include <hip/hip_runtime.h>
#include <hip/hip_bf16.h>
#include <math.h>
#include <stdint.h>

// ---------------------------------------------------------------------------
// GATv2 (3 layers, 4 heads x 64ch) + segment-max readout + MLP head, gfx1250.
// All GEMMs in bf16 WMMA 16x16x32 (fp32 accumulate) fed by pre-converted
// bf16 operands (activations row-major, weights transposed) so every fragment
// is two contiguous b128 loads. Edge tile staged to LDS via async-to-LDS DMA.
// ---------------------------------------------------------------------------

typedef __attribute__((ext_vector_type(16))) __bf16 v16bf;
typedef __attribute__((ext_vector_type(8)))  __bf16 v8bf;
typedef __attribute__((ext_vector_type(8)))  float  v8f;
typedef __bf16 bf16_t;

#define HID       64
#define HC        256          // HEADS*HID
#define HEADS     4
#define EDGE_F    60
#define EF_PAD    64           // EDGE_F padded to WMMA K granularity
#define NGRAPH    16
#define OUT_DIM   1800
#define NEG_SLOPE 0.2f
#define SIGMA     2.0f         // R_CUT/3
#define BN_EPS    1e-5f

__device__ __forceinline__ v16bf cat8(v8bf lo, v8bf hi) {
  return __builtin_shufflevector(lo, hi, 0, 1, 2, 3, 4, 5, 6, 7,
                                 8, 9, 10, 11, 12, 13, 14, 15);
}

// ---- monotone float <-> uint encoding for atomicMax on floats --------------
__device__ __forceinline__ unsigned int enc_f32(float f) {
  unsigned int u = __float_as_uint(f);
  return (u & 0x80000000u) ? ~u : (u | 0x80000000u);
}
__device__ __forceinline__ float dec_f32(unsigned int u) {
  u = (u & 0x80000000u) ? (u & 0x7FFFFFFFu) : ~u;
  return __uint_as_float(u);
}

// ---------------------------------------------------------------------------
// K1: gaussian-smeared edge attributes (bf16, padded K=64) + degree +
//     loop-attr accumulation. One block (64 threads) per edge.
// ---------------------------------------------------------------------------
__global__ void edge_attr_kernel(const float* __restrict__ evec,
                                 const int*   __restrict__ ei,
                                 int E,
                                 bf16_t* __restrict__ ea2b,
                                 float* __restrict__ loop_sum,
                                 float* __restrict__ deg) {
  int e = blockIdx.x;
  int f = threadIdx.x;                 // 0..63
  if (e >= E) return;
  float vx = evec[e * 3 + 0], vy = evec[e * 3 + 1], vz = evec[e * 3 + 2];
  float el = sqrtf(vx * vx + vy * vy + vz * vz);
  int t = ei[E + e];
  if (f < EDGE_F) {
    float d   = el - 0.1f * (float)f;
    float val = __expf(-0.5f * d * d / (SIGMA * SIGMA)) *
                (1.0f / (SIGMA * 2.5066282746310002f));
    ea2b[(size_t)e * EF_PAD + f] = (bf16_t)val;
    atomicAdd(&loop_sum[(size_t)t * EDGE_F + f], val);
  } else {
    ea2b[(size_t)e * EF_PAD + f] = (bf16_t)0.0f;     // K padding
  }
  if (f == 0) atomicAdd(&deg[t], 1.0f);
}

// K2: self-loop attributes ('mean' fill) appended after the E real edges
__global__ void loop_attr_kernel(const float* __restrict__ ls,
                                 const float* __restrict__ deg,
                                 int N, int E,
                                 bf16_t* __restrict__ ea2b) {
  int idx = blockIdx.x * blockDim.x + threadIdx.x;
  if (idx >= N * EF_PAD) return;
  int n = idx >> 6, f = idx & 63;
  float v = (f < EDGE_F) ? ls[(size_t)n * EDGE_F + f] / fmaxf(deg[n], 1.0f)
                         : 0.0f;
  ea2b[(size_t)(E + n) * EF_PAD + f] = (bf16_t)v;
}

// ---- bf16 staging kernels --------------------------------------------------
__global__ void cvt_bf16(const float* __restrict__ in,
                         bf16_t* __restrict__ out, int count) {
  int idx = blockIdx.x * blockDim.x + threadIdx.x;
  if (idx < count) out[idx] = (bf16_t)in[idx];
}

// Wt[n][k] = W[k][n] (bf16, zero-padded to Kpad)
__global__ void transpose_w_bf16(const float* __restrict__ W,
                                 int K, int Kpad,
                                 bf16_t* __restrict__ Wt) {
  int idx = blockIdx.x * blockDim.x + threadIdx.x;
  if (idx >= HC * Kpad) return;
  int n = idx / Kpad, k = idx % Kpad;
  Wt[idx] = (k < K) ? (bf16_t)W[(size_t)k * HC + n] : (bf16_t)0.0f;
}

// ---------------------------------------------------------------------------
// K3: node GEMM  Y[M,256] = Xb[M,K] @ Wt[256,K]^T + bias
// bf16 WMMA; one wave32 per 16x16 output tile; every fragment = 2 x b128.
// ---------------------------------------------------------------------------
__global__ void gemm_node_wmma(const bf16_t* __restrict__ Xb,  // [M][K]
                               const bf16_t* __restrict__ Wt,  // [256][K]
                               const float*  __restrict__ bias,
                               float* __restrict__ Y,
                               int M, int K) {
  int gwave = (blockIdx.x * blockDim.x + threadIdx.x) >> 5;
  int lane  = threadIdx.x & 31;
  int tm = gwave >> 4;             // 16 N-tiles cover 256 cols
  int tn = gwave & 15;
  int mtiles = (M + 15) >> 4;
  if (tm >= mtiles) return;        // wave-uniform: EXEC stays all-1s

  int lrow = lane & 15;
  int m = tm * 16 + lrow;  if (m >= M) m = M - 1;
  int n = tn * 16 + lrow;
  const bf16_t* arow = Xb + (size_t)m * K;
  const bf16_t* brow = Wt + (size_t)n * K;
  int asel = (lane < 16) ? 0 : 8;   // A lane layout (ISA 7.12.2, 16-bit 16x32)
  int bsel = (lane < 16) ? 0 : 8;   // B: same K layout as A

  v8f acc = {};
  for (int k0 = 0; k0 < K; k0 += 32) {
    v8bf alo = *(const v8bf*)(arow + k0 + asel);
    v8bf ahi = *(const v8bf*)(arow + k0 + 16 + asel);
    v8bf blo = *(const v8bf*)(brow + k0 + bsel);
    v8bf bhi = *(const v8bf*)(brow + k0 + 16 + bsel);
    if (k0 + 32 < K) __builtin_prefetch(arow + k0 + 32, 0, 3); // global_prefetch
    acc = __builtin_amdgcn_wmma_f32_16x16x32_bf16(false, cat8(alo, ahi),
                                                  false, cat8(blo, bhi),
                                                  (short)0, acc, false, false);
  }
  float bn = bias[n];
  if (tm * 16 + 16 <= M) {          // full-tile fast path (M % 16 == 0 here)
#pragma unroll
    for (int r = 0; r < 8; ++r) {
      int mr = tm * 16 + ((lane < 16) ? r : r + 8);
      Y[(size_t)mr * HC + n] = acc[r] + bn;
    }
  } else {
#pragma unroll
    for (int r = 0; r < 8; ++r) {
      int mr = tm * 16 + ((lane < 16) ? r : r + 8);
      if (mr < M) Y[(size_t)mr * HC + n] = acc[r] + bn;
    }
  }
}

// ---------------------------------------------------------------------------
// K4: fused edge kernel. Per block: 16 edges.
//   Stage ea tile (16x64 bf16) to LDS via async-to-LDS DMA (ASYNCcnt path)
//   P = ea_tile @ We (WMMA, 8 waves x 2 N-tiles); m = P + xl[src] + xr[tgt]
//   logit[edge,h] = sum_c leaky_relu(m)*att[h,c]  (LDS atomic reduction)
//   emax[tgt,h]   = atomicMax(logit)
// ---------------------------------------------------------------------------
__global__ void edge_logits_wmma(const bf16_t* __restrict__ ea2b, // [E2][64]
                                 const bf16_t* __restrict__ WeT,  // [256][64]
                                 const float*  __restrict__ xl,
                                 const float*  __restrict__ xr,
                                 const float*  __restrict__ att,
                                 const int*    __restrict__ ei,
                                 int E, int E2,
                                 float* __restrict__ elog,
                                 unsigned int* __restrict__ emax_u) {
  __shared__ bf16_t sAb[16][EF_PAD];         // 2 KB
  __shared__ int    ss[16], tt[16];
  __shared__ float  slog[16 * HEADS];

  int tile = blockIdx.x;
  int tid  = threadIdx.x;
  int e0   = tile * 16;

  // ---- async DMA: 256 threads x 8B = the whole 16x128B tile ----
  {
    int r = tid >> 4, chunk = tid & 15;      // 16 chunks of 8B per row
    int er = e0 + r; if (er >= E2) er = E2 - 1;
    const bf16_t* gp = ea2b + (size_t)er * EF_PAD + chunk * 4;
    unsigned int lds = (unsigned int)(uintptr_t)(&sAb[r][chunk * 4]);
    asm volatile("global_load_async_to_lds_b64 %0, %1, off"
                 :: "v"(lds), "v"(gp) : "memory");
  }
  if (tid < 16) {                            // stage src/tgt
    int er = e0 + tid; if (er >= E2) er = E2 - 1;
    int s, t;
    if (er < E) { s = ei[er]; t = ei[E + er]; } else { s = t = er - E; }
    ss[tid] = s; tt[tid] = t;
  }
  if (tid < 16 * HEADS) slog[tid] = 0.0f;
  asm volatile("s_wait_asynccnt 0" ::: "memory");
  __syncthreads();

  int wave = tid >> 5;
  int lane = tid & 31;
  int lrow = lane & 15;
  int asel = (lane < 16) ? 0 : 8;
  int bsel = (lane < 16) ? 0 : 8;

  for (int nt = wave; nt < 16; nt += 8) {    // 8 waves cover 16 N-tiles
    int n = nt * 16 + lrow;                  // channel 0..255
    const bf16_t* brow = WeT + (size_t)n * EF_PAD;
    v8f acc = {};
#pragma unroll
    for (int k0 = 0; k0 < EF_PAD; k0 += 32) {
      v8bf alo = *(const v8bf*)&sAb[lrow][k0 + asel];        // ds_load_b128
      v8bf ahi = *(const v8bf*)&sAb[lrow][k0 + 16 + asel];
      v8bf blo = *(const v8bf*)(brow + k0 + bsel);
      v8bf bhi = *(const v8bf*)(brow + k0 + 16 + bsel);
      acc = __builtin_amdgcn_wmma_f32_16x16x32_bf16(false, cat8(alo, ahi),
                                                    false, cat8(blo, bhi),
                                                    (short)0, acc, false, false);
    }
    int   head = n >> 6;
    float aw   = att[head * HID + (n & 63)];
#pragma unroll
    for (int r = 0; r < 8; ++r) {
      int row = (lane < 16) ? r : (r + 8);
      int er  = e0 + row;
      if (er < E2) {
        int s = ss[row], t = tt[row];
        float m = acc[r] + xl[(size_t)s * HC + n] + xr[(size_t)t * HC + n];
        float g = (m > 0.0f) ? m : NEG_SLOPE * m;
        atomicAdd(&slog[row * HEADS + head], g * aw);        // ds_add_f32
      }
    }
  }
  __syncthreads();
  if (tid < 16 * HEADS) {
    int row = tid >> 2, h = tid & 3;
    int er  = e0 + row;
    if (er < E2) {
      float lg = slog[tid];
      elog[(size_t)er * HEADS + h] = lg;
      atomicMax(&emax_u[tt[row] * HEADS + h], enc_f32(lg));
    }
  }
}

// K5a: a = exp(e - emax[tgt]); denom[tgt] += a
__global__ void softmax_norm(float* __restrict__ elog,
                             const unsigned int* __restrict__ emax_u,
                             float* __restrict__ denom,
                             const int* __restrict__ ei,
                             int E, int E2) {
  int idx = blockIdx.x * blockDim.x + threadIdx.x;
  if (idx >= E2 * HEADS) return;
  int er = idx >> 2, h = idx & 3;
  int t  = (er < E) ? ei[E + er] : (er - E);
  float mx = dec_f32(emax_u[t * HEADS + h]);
  float a  = __expf(elog[idx] - mx);
  elog[idx] = a;
  atomicAdd(&denom[t * HEADS + h], a);
}

// K5b: out[tgt] += (a/denom[tgt]) * xl[src]   — one block per edge, thread=ch
__global__ void aggregate_kernel(const float* __restrict__ elog,
                                 const float* __restrict__ denom,
                                 const float* __restrict__ xl,
                                 const int*   __restrict__ ei,
                                 int E, int E2,
                                 float* __restrict__ outb) {
  int er = blockIdx.x;
  int c  = threadIdx.x;
  if (er >= E2) return;
  int s, t;
  if (er < E) { s = ei[er]; t = ei[E + er]; } else { s = t = er - E; }
  int h = c >> 6;
  float alpha = elog[(size_t)er * HEADS + h] / denom[t * HEADS + h];
  atomicAdd(&outb[(size_t)t * HC + c], alpha * xl[(size_t)s * HC + c]);
}

// K6: h = relu(out + bias); segment-max into per-graph feature (encoded)
__global__ void finish_layer(const float* __restrict__ outb,
                             const float* __restrict__ bias,
                             const int*   __restrict__ batch,
                             int N,
                             float* __restrict__ hbuf,
                             unsigned int* __restrict__ gfmax_u) {
  int idx = blockIdx.x * blockDim.x + threadIdx.x;
  if (idx >= N * HC) return;
  int n = idx / HC, c = idx % HC;
  float v = fmaxf(outb[idx] + bias[c], 0.0f);
  hbuf[idx] = v;
  atomicMax(&gfmax_u[batch[n] * HC + c], enc_f32(v));
}

// K7: global_feature += decoded segment max
__global__ void gf_add(const unsigned int* __restrict__ gfmax_u,
                       float* __restrict__ gf_accum) {
  int idx = blockIdx.x * blockDim.x + threadIdx.x;
  if (idx < NGRAPH * HC) gf_accum[idx] += dec_f32(gfmax_u[idx]);
}

// K8a: gf1 = gf@lin1 + b; BN over the 16 graphs (biased var); relu
__global__ void mlp1_bn(const float* __restrict__ gf,
                        const float* __restrict__ w1,
                        const float* __restrict__ b1,
                        const float* __restrict__ bng,
                        const float* __restrict__ bnb,
                        float* __restrict__ gf2) {
  int j = threadIdx.x;                       // 0..255
  float v[NGRAPH];
#pragma unroll 1
  for (int g = 0; g < NGRAPH; ++g) {
    float s = b1[j];
    for (int k = 0; k < HC; ++k) s += gf[g * HC + k] * w1[k * HC + j];
    v[g] = s;
  }
  float mu = 0.0f;
#pragma unroll
  for (int g = 0; g < NGRAPH; ++g) mu += v[g];
  mu *= (1.0f / NGRAPH);
  float var = 0.0f;
#pragma unroll
  for (int g = 0; g < NGRAPH; ++g) { float d = v[g] - mu; var += d * d; }
  var *= (1.0f / NGRAPH);
  float inv = rsqrtf(var + BN_EPS);
#pragma unroll
  for (int g = 0; g < NGRAPH; ++g) {
    float t = (v[g] - mu) * inv * bng[j] + bnb[j];
    gf2[g * HC + j] = fmaxf(t, 0.0f);
  }
}

// K8b: out[16,1800] = gf2 @ lin2 + b2
__global__ void mlp2(const float* __restrict__ gf2,
                     const float* __restrict__ w2,
                     const float* __restrict__ b2,
                     float* __restrict__ out) {
  int idx = blockIdx.x * blockDim.x + threadIdx.x;
  if (idx >= NGRAPH * OUT_DIM) return;
  int g = idx / OUT_DIM, o = idx % OUT_DIM;
  float s = b2[o];
  for (int k = 0; k < HC; ++k) s += gf2[g * HC + k] * w2[k * OUT_DIM + o];
  out[idx] = s;
}

// ---------------------------------------------------------------------------
extern "C" void kernel_launch(void* const* d_in, const int* in_sizes, int n_in,
                              void* d_out, int out_size, void* d_ws, size_t ws_size,
                              hipStream_t stream) {
  const float* x     = (const float*)d_in[0];
  const int*   ei    = (const int*)  d_in[1];   // [2,E]: row0 src, row1 tgt
  const float* evec  = (const float*)d_in[2];
  const int*   batch = (const int*)  d_in[3];

  const int N  = in_sizes[0] / HID;
  const int E  = in_sizes[1] / 2;
  const int E2 = E + N;

  // layer params: (Wl, bl, Wr, br, We, att, bias) x 3 — flattened leaves or
  // a single concatenated blob, depending on harness pytree handling.
  const float* P[3][7];
  if (n_in >= 25) {
    for (int l = 0; l < 3; ++l)
      for (int j = 0; j < 7; ++j)
        P[l][j] = (const float*)d_in[4 + 7 * l + j];
  } else {
    const float* base = (const float*)d_in[4];
    size_t poff = 0;
    int in_ch = HID;
    for (int l = 0; l < 3; ++l) {
      size_t sz[7] = {(size_t)in_ch * HC, HC, (size_t)in_ch * HC, HC,
                      (size_t)EDGE_F * HC, (size_t)HEADS * HID, HC};
      for (int j = 0; j < 7; ++j) { P[l][j] = base + poff; poff += sz[j]; }
      in_ch = HC;
    }
  }
  const float* lin1_w = (const float*)d_in[n_in - 6];
  const float* lin1_b = (const float*)d_in[n_in - 5];
  const float* bn_g   = (const float*)d_in[n_in - 4];
  const float* bn_b   = (const float*)d_in[n_in - 3];
  const float* lin2_w = (const float*)d_in[n_in - 2];
  const float* lin2_b = (const float*)d_in[n_in - 1];

  // ---- carve workspace (deterministic every call) ----
  char* ws  = (char*)d_ws;
  size_t off = 0;
  auto alloc = [&](size_t bytes) -> char* {
    char* p = ws + off;
    off += (bytes + 255) & ~(size_t)255;
    return p;
  };
  bf16_t* ea2b     = (bf16_t*)alloc((size_t)E2 * EF_PAD * 2);
  float*  loop_sum = (float*) alloc((size_t)N * EDGE_F * 4);
  float*  deg      = (float*) alloc((size_t)N * 4);
  bf16_t* xb       = (bf16_t*)alloc((size_t)N * HC * 2);   // bf16 activations
  bf16_t* WtL      = (bf16_t*)alloc((size_t)HC * HC * 2);  // Wl^T
  bf16_t* WtR      = (bf16_t*)alloc((size_t)HC * HC * 2);  // Wr^T
  bf16_t* WeT      = (bf16_t*)alloc((size_t)HC * EF_PAD * 2);
  float*  xl       = (float*) alloc((size_t)N * HC * 4);
  float*  xr       = (float*) alloc((size_t)N * HC * 4);
  float*  hbuf     = (float*) alloc((size_t)N * HC * 4);
  float*  outb     = (float*) alloc((size_t)N * HC * 4);
  float*  elog     = (float*) alloc((size_t)E2 * HEADS * 4);
  unsigned int* emax_u  = (unsigned int*)alloc((size_t)N * HEADS * 4);
  float*  denom    = (float*) alloc((size_t)N * HEADS * 4);
  unsigned int* gfmax_u = (unsigned int*)alloc((size_t)NGRAPH * HC * 4);
  float*  gf_accum = (float*) alloc((size_t)NGRAPH * HC * 4);
  float*  gf2      = (float*) alloc((size_t)NGRAPH * HC * 4);
  (void)ws_size; (void)out_size;

  // ---- edge attributes + self-loop fill-value ('mean') ----
  hipMemsetAsync(loop_sum, 0, (size_t)N * EDGE_F * 4, stream);
  hipMemsetAsync(deg,      0, (size_t)N * 4,          stream);
  hipMemsetAsync(gf_accum, 0, (size_t)NGRAPH * HC * 4, stream);

  edge_attr_kernel<<<E, EF_PAD, 0, stream>>>(evec, ei, E, ea2b, loop_sum, deg);
  loop_attr_kernel<<<(N * EF_PAD + 255) / 256, 256, 0, stream>>>(
      loop_sum, deg, N, E, ea2b);

  // ---- GATv2 layers ----
  for (int l = 0; l < 3; ++l) {
    const float* hin = (l == 0) ? x : hbuf;
    int K_in = (l == 0) ? HID : HC;

    // bf16 staging: activations + transposed weights
    cvt_bf16<<<((size_t)N * K_in + 255) / 256, 256, 0, stream>>>(hin, xb, N * K_in);
    transpose_w_bf16<<<(HC * K_in + 255) / 256, 256, 0, stream>>>(P[l][0], K_in, K_in, WtL);
    transpose_w_bf16<<<(HC * K_in + 255) / 256, 256, 0, stream>>>(P[l][2], K_in, K_in, WtR);
    transpose_w_bf16<<<(HC * EF_PAD + 255) / 256, 256, 0, stream>>>(P[l][4], EDGE_F, EF_PAD, WeT);

    int mtiles = (N + 15) / 16;
    int blocks = (mtiles * 16 * 32 + 255) / 256;     // waves = mtiles * 16
    gemm_node_wmma<<<blocks, 256, 0, stream>>>(xb, WtL, P[l][1], xl, N, K_in);
    gemm_node_wmma<<<blocks, 256, 0, stream>>>(xb, WtR, P[l][3], xr, N, K_in);

    hipMemsetAsync(emax_u, 0, (size_t)N * HEADS * 4, stream);   // enc(very small)
    hipMemsetAsync(denom,  0, (size_t)N * HEADS * 4, stream);
    hipMemsetAsync(outb,   0, (size_t)N * HC * 4,    stream);
    hipMemsetAsync(gfmax_u,0, (size_t)NGRAPH * HC * 4, stream);

    edge_logits_wmma<<<(E2 + 15) / 16, 256, 0, stream>>>(
        ea2b, WeT, xl, xr, P[l][5], ei, E, E2, elog, emax_u);

    softmax_norm<<<((size_t)E2 * HEADS + 255) / 256, 256, 0, stream>>>(
        elog, emax_u, denom, ei, E, E2);

    aggregate_kernel<<<E2, HC, 0, stream>>>(elog, denom, xl, ei, E, E2, outb);

    finish_layer<<<((size_t)N * HC + 255) / 256, 256, 0, stream>>>(
        outb, P[l][6], batch, N, hbuf, gfmax_u);

    gf_add<<<(NGRAPH * HC + 255) / 256, 256, 0, stream>>>(gfmax_u, gf_accum);
  }

  // ---- MLP head ----
  mlp1_bn<<<1, HC, 0, stream>>>(gf_accum, lin1_w, lin1_b, bn_g, bn_b, gf2);
  mlp2<<<(NGRAPH * OUT_DIM + 255) / 256, 256, 0, stream>>>(
      gf2, lin2_w, lin2_b, (float*)d_out);
}
